// RNNModel_4423816315187
// MI455X (gfx1250) — matrix operand order repaired and checked by reading, not compiled, not run on hardware
//
#include <hip/hip_runtime.h>

typedef _Float16 half_t;
typedef __attribute__((ext_vector_type(16))) _Float16 v16h;
typedef __attribute__((ext_vector_type(8)))  _Float16 v8h;
typedef __attribute__((ext_vector_type(8)))  float    v8f;

namespace {
constexpr int BATCH = 8192;
constexpr int T     = 1024;
constexpr int I     = 7;
constexpr int H     = 100;
constexpr int KP    = 128;   // padded K: h[0..99], x[100..106], bias-1[107], zeros[108..127]
constexpr int LDA   = 136;   // LDS row stride (elements); 272B dodges 256B bank stride
constexpr int ROWS  = 32;    // batch rows per block
constexpr int NTHREADS = 128; // 4 wave32
constexpr int PF    = 8;     // x-load software pipeline depth (steps of latency cover)
}

static __device__ inline v8f wmma16(v16h a, v16h b, v8f c) {
  // D = A(16x32 f16) * B(32x16 f16) + C(f32)
  return __builtin_amdgcn_wmma_f32_16x16x32_f16(false, a, false, b, (short)0, c,
                                                false, false);
}

static __device__ inline float relu(float v) {
  // med3(x, 0, +inf) == max(0, x): single v_med3_f32, no canonicalize pair
  return __builtin_amdgcn_fmed3f(v, 0.0f, __builtin_inff());
}

// A 16x32 f16 layout: lanes 0-15 -> row M=lane, K=[k0..k0+7] in elems 0..7 and
// K=[k0+16..k0+23] in elems 8..15; lanes 16-31 -> same rows, k0 += 8.
static __device__ inline v16h ld_frag_A(const half_t* base, int lane, int m, int kt) {
  const int r  = m * 16 + (lane & 15);
  const int k0 = kt * 32 + ((lane >> 4) << 3);
  const v8h lo = *(const v8h*)(base + r * LDA + k0);
  const v8h hi = *(const v8h*)(base + r * LDA + k0 + 16);
  v16h f;
#pragma unroll
  for (int i = 0; i < 8; ++i) { f[i] = lo[i]; f[i + 8] = hi[i]; }
  return f;
}

// B 32x16 f16 layout: lane holds column n = lane%16; lanes 0-15 -> K=k0..k0+7 in
// elems 0..7 and K=k0+16..k0+23 in elems 8..15; lanes 16-31 -> same, k0 += 8.
static __device__ inline v16h ld_frag_B(const half_t* base, int lane, int nt, int kt) {
  const int n  = nt * 16 + (lane & 15);
  const int k0 = kt * 32 + ((lane >> 4) << 3);
  const v8h lo = *(const v8h*)(base + n * LDA + k0);
  const v8h hi = *(const v8h*)(base + n * LDA + k0 + 16);
  v16h f;
#pragma unroll
  for (int i = 0; i < 8; ++i) { f[i] = lo[i]; f[i + 8] = hi[i]; }
  return f;
}

__global__ __launch_bounds__(NTHREADS) void rnn_relu_wmma(
    const float* __restrict__ x,    const float* __restrict__ W_ih,
    const float* __restrict__ W_hh, const float* __restrict__ b_ih,
    const float* __restrict__ b_hh, const float* __restrict__ W_fc,
    const float* __restrict__ b_fc, float* __restrict__ out) {
  __shared__ __align__(16) half_t Wl[KP * LDA];          // W_aug, row n over k
  __shared__ __align__(16) half_t gbuf[2][ROWS * LDA];   // augmented state g

  const int tid     = threadIdx.x;
  const int lane    = tid & 31;
  const int wave    = tid >> 5;
  const int rowBase = blockIdx.x * ROWS;

  // ---- Build augmented weight matrix W_aug[n][k] in LDS ----
  for (int idx = tid; idx < KP * LDA; idx += NTHREADS) {
    const int n = idx / LDA, k = idx - n * LDA;
    float v = 0.0f;
    if (n < H && k < KP) {
      if (k < H)           v = W_hh[n * H + k];
      else if (k < H + I)  v = W_ih[n * I + (k - H)];
      else if (k == H + I) v = b_ih[n] + b_hh[n];
    }
    Wl[idx] = (half_t)v;
  }
  for (int idx = tid; idx < ROWS * LDA; idx += NTHREADS) {
    gbuf[0][idx] = (half_t)0.0f;
    gbuf[1][idx] = (half_t)0.0f;
  }
  __syncthreads();

  // ---- Persistent B fragments: wave w owns N-tiles 2w, 2w+1.
  // Wave 3's tile 7 covers the zero-pad rows of W_aug: WMMAs produce zeros and
  // its stores are masked, so every wave runs an identical branch-free body.
  v16h Bf[2][4];
#pragma unroll
  for (int kt = 0; kt < 4; ++kt) {
    Bf[0][kt] = ld_frag_B(Wl, lane, 2 * wave, kt);
    Bf[1][kt] = ld_frag_B(Wl, lane, 2 * wave + 1, kt);
  }

  // ---- Per-thread x stream: 2 slots (rows r0, r0+16; column i0).
  // Clamp i0 so loads are UNCONDITIONAL (no exec-masked loads => the scheduler
  // can keep them in flight); the i0==7 slot becomes the constant-1 bias at
  // store time.
  const int r0 = tid >> 3, i0 = tid & 7;
  const int ic = (i0 < I) ? i0 : (I - 1);
  const float* xp0 = x + (size_t)(rowBase + r0)      * (T * I) + ic;
  const float* xp1 = x + (size_t)(rowBase + r0 + 16) * (T * I) + ic;
  const int xoff0 = r0 * LDA + H + i0;
  const int xoff1 = (r0 + 16) * LDA + H + i0;

  // ---- g_0: h=0 (already), x_0 at k=100..106, 1.0 at k=107 ----
  gbuf[0][xoff0] = (i0 < I) ? (half_t)xp0[0] : (half_t)1.0f;
  gbuf[0][xoff1] = (i0 < I) ? (half_t)xp1[0] : (half_t)1.0f;
  __syncthreads();

  // ---- x-load register FIFO: xpipe[k] holds x_{t+k+1} at iteration base t.
  // PF steps of in-flight distance (~8 step bodies) covers an HBM miss.
  float xpipe[PF][2];
#pragma unroll
  for (int k = 0; k < PF; ++k) {
    const int tn = (k + 1 < T) ? (k + 1) : (T - 1);
    xpipe[k][0] = xp0[tn * I];
    xpipe[k][1] = xp1[tn * I];
  }

  auto step = [&](const half_t* gp, half_t* gq, int t, int k) {
    // Consume the pipelined x_{t+1}, then refill slot k with x_{t+1+PF}.
    const float xv0 = xpipe[k][0];
    const float xv1 = xpipe[k][1];
    {
      const int tn = (t + 1 + PF < T) ? (t + 1 + PF) : (T - 1);
      xpipe[k][0] = xp0[tn * I];
      xpipe[k][1] = xp1[tn * I];
    }

    // Hoist all A-fragment loads, then the 16-WMMA chain (4 independent
    // 4-deep accumulation chains; DS batch drains underneath).
    v16h Af[4][2];
#pragma unroll
    for (int kt = 0; kt < 4; ++kt) {
      Af[kt][0] = ld_frag_A(gp, lane, 0, kt);
      Af[kt][1] = ld_frag_A(gp, lane, 1, kt);
    }

    const v8f vz = {0, 0, 0, 0, 0, 0, 0, 0};
    v8f acc[2][2] = {{vz, vz}, {vz, vz}};
#pragma unroll
    for (int kt = 0; kt < 4; ++kt) {
      acc[0][0] = wmma16(Af[kt][0], Bf[0][kt], acc[0][0]);
      acc[1][0] = wmma16(Af[kt][1], Bf[0][kt], acc[1][0]);
      acc[0][1] = wmma16(Af[kt][0], Bf[1][kt], acc[0][1]);
      acc[1][1] = wmma16(Af[kt][1], Bf[1][kt], acc[1][1]);
    }

    // relu -> f16 -> store h_{t+1}; C/D layout: vgpr v holds rows m*16+v(+8)
#pragma unroll
    for (int j = 0; j < 2; ++j) {
      const int n = (2 * wave + j) * 16 + (lane & 15);
      if (n < H) {
#pragma unroll
        for (int m = 0; m < 2; ++m) {
          const int rb = m * 16 + ((lane >> 4) << 3);
#pragma unroll
          for (int v = 0; v < 8; ++v) {
            gq[(rb + v) * LDA + n] = (half_t)relu(acc[m][j][v]);
          }
        }
      }
    }

    // x_{t+1} and the constant-1 bias slot into the next buffer
    gq[xoff0] = (i0 < I) ? (half_t)xv0 : (half_t)1.0f;
    gq[xoff1] = (i0 < I) ? (half_t)xv1 : (half_t)1.0f;
    __syncthreads();
  };

  for (int t = 0; t < T; t += PF) {
#pragma unroll
    for (int k = 0; k < PF; k += 2) {
      step(gbuf[0], gbuf[1], t + k, k);
      step(gbuf[1], gbuf[0], t + k + 1, k + 1);
    }
  }

  // ---- FC head: out[b] = h_T[b,:] . W_fc + b_fc (h_T is in gbuf[0], T even) ----
  if (tid < ROWS) {
    float a = b_fc[0];
#pragma unroll 4
    for (int kk = 0; kk < H; ++kk)
      a += (float)gbuf[0][tid * LDA + kk] * W_fc[kk];
    out[rowBase + tid] = a;
  }
}

extern "C" void kernel_launch(void* const* d_in, const int* in_sizes, int n_in,
                              void* d_out, int out_size, void* d_ws, size_t ws_size,
                              hipStream_t stream) {
  const float* xp   = (const float*)d_in[0];
  const float* Wih  = (const float*)d_in[1];
  const float* Whh  = (const float*)d_in[2];
  const float* bih  = (const float*)d_in[3];
  const float* bhh  = (const float*)d_in[4];
  const float* Wfc  = (const float*)d_in[5];
  const float* bfc  = (const float*)d_in[6];
  float* out = (float*)d_out;

  dim3 grid(BATCH / ROWS), block(NTHREADS);
  hipLaunchKernelGGL(rnn_relu_wmma, grid, block, 0, stream,
                     xp, Wih, Whh, bih, bhh, Wfc, bfc, out);
}
